// Qwen3MoeRouter_58317065945109
// MI455X (gfx1250) — matrix operand order repaired and checked by reading, not compiled, not run on hardware
//
#include <hip/hip_runtime.h>
#include <hip/hip_bf16.h>

typedef __attribute__((ext_vector_type(16))) _Float16 v16h;
typedef __attribute__((ext_vector_type(8)))  float    v8f;
typedef __attribute__((ext_vector_type(4)))  float    f32x4;

#define NUM_TOKENS  131072
#define HIDDEN      2048
#define NUM_EXPERTS 128
#define TOP_K       8
#define KSTEPS      (HIDDEN / 32)      // 64 k-steps of 32
#define NTILES      (NUM_EXPERTS / 16) // 8 expert tiles of 16
#define TOK_PER_WG  128                // 4 waves x 32 tokens

// ---------------------------------------------------------------------------
// Pack gate_w [E=128, H=2048] fp32 into f16 B-fragments in the exact CDNA5
// WMMA B (32x16, f16) lane layout (ISA 7.12.2):
//   lane 0-15  hold column N = lane,    halves j=0..7 -> K = j, j=8..15 -> K = 16+(j-8)
//   lane 16-31 hold column N = lane-16, same with K offset +8
// ws layout (v16h units): frag[((nt*KSTEPS + kstep)*32 + lane)]
// Total: 8*64*32*16 halves = 512 KB (gate_w itself is only 1 MB).
// ---------------------------------------------------------------------------
__global__ void gate_pack_f16(const float* __restrict__ gate_w,
                              _Float16* __restrict__ bws) {
    int idx   = blockIdx.x * blockDim.x + threadIdx.x;   // 0 .. 262143
    int j     = idx & 15;
    int lane  = (idx >> 4) & 31;
    int kstep = (idx >> 9) & 63;
    int nt    = idx >> 15;
    int e = nt * 16 + (lane & 15);
    int k = kstep * 32 + ((j < 8) ? j : j + 8) + ((lane >= 16) ? 8 : 0);
    bws[idx] = (_Float16)gate_w[e * HIDDEN + k];
}

// ---------------------------------------------------------------------------
// Fused router: WMMA GEMM (f16 in / f32 acc) + top-8 + renormalized weights.
// 128 threads = 4 waves; each wave: 32 tokens (2 M-tiles) x 128 experts,
// so every B fragment feeds TWO WMMAs (halves first-level-cache traffic).
// Workgroup: 128 tokens. Grid: NUM_TOKENS/128 = 1024 workgroups.
// ---------------------------------------------------------------------------
__launch_bounds__(128)
__global__ void router_fused(const float* __restrict__ hs,
                             const _Float16* __restrict__ bws,
                             float* __restrict__ out_w,
                             int* __restrict__ out_i) {
    __shared__ float logits[TOK_PER_WG * 129];   // odd stride -> conflict-free

    const int lane  = threadIdx.x & 31;
    const int wave  = threadIdx.x >> 5;
    const int mbase = blockIdx.x * TOK_PER_WG + wave * 32;

    // --- WMMA A addressing (16x32 f16, ISA 7.12.2) ---
    // lane 0-15: M=lane,  halves j=0..7 -> K=j, j=8..15 -> K=16+(j-8)
    // lane 16-31: same M rows, K offset +8 inside each half-group
    const int   mrow  = mbase + (lane & 15);
    const int   kbias = (lane >= 16) ? 8 : 0;
    const float* arow0 = hs + (size_t)mrow * HIDDEN + kbias;            // M-tile 0
    const float* arow1 = arow0 + (size_t)16 * HIDDEN;                   // M-tile 1

    v8f c0[NTILES], c1[NTILES];
#pragma unroll
    for (int nt = 0; nt < NTILES; ++nt)
#pragma unroll
        for (int i = 0; i < 8; ++i) { c0[nt][i] = 0.0f; c1[nt][i] = 0.0f; }

    const v16h* bfrag = (const v16h*)bws + lane;

    for (int kstep = 0; kstep < KSTEPS; ++kstep) {
        const float* ak0 = arow0 + kstep * 32;
        const float* ak1 = arow1 + kstep * 32;
        f32x4 p0 = *(const f32x4*)(ak0 + 0);
        f32x4 p1 = *(const f32x4*)(ak0 + 4);
        f32x4 p2 = *(const f32x4*)(ak0 + 16);
        f32x4 p3 = *(const f32x4*)(ak0 + 20);
        f32x4 q0 = *(const f32x4*)(ak1 + 0);
        f32x4 q1 = *(const f32x4*)(ak1 + 4);
        f32x4 q2 = *(const f32x4*)(ak1 + 16);
        f32x4 q3 = *(const f32x4*)(ak1 + 20);
        v16h a0, a1;
#pragma unroll
        for (int i = 0; i < 4; ++i) {
            a0[i]      = (_Float16)p0[i];
            a0[4 + i]  = (_Float16)p1[i];
            a0[8 + i]  = (_Float16)p2[i];
            a0[12 + i] = (_Float16)p3[i];
            a1[i]      = (_Float16)q0[i];
            a1[4 + i]  = (_Float16)q1[i];
            a1[8 + i]  = (_Float16)q2[i];
            a1[12 + i] = (_Float16)q3[i];
        }
#pragma unroll
        for (int nt = 0; nt < NTILES; ++nt) {
            v16h b = bfrag[(nt * KSTEPS + kstep) * 32];
            c0[nt] = __builtin_amdgcn_wmma_f32_16x16x32_f16(
                false, a0, false, b, (short)0, c0[nt], false, false);
            c1[nt] = __builtin_amdgcn_wmma_f32_16x16x32_f16(
                false, a1, false, b, (short)0, c1[nt], false, false);
        }
    }

    // --- spill accumulators to LDS ---
    // C/D 16x16 f32: VGPR r, lane l -> M = r + (l>=16 ? 8 : 0), N = l%16
    const int thalf = (lane >= 16) ? 8 : 0;
    const int ncol  = lane & 15;
#pragma unroll
    for (int nt = 0; nt < NTILES; ++nt)
#pragma unroll
        for (int r = 0; r < 8; ++r) {
            logits[(wave * 32 + r + thalf) * 129 + nt * 16 + ncol]      = c0[nt][r];
            logits[(wave * 32 + 16 + r + thalf) * 129 + nt * 16 + ncol] = c1[nt][r];
        }

    __syncthreads();

    // --- per-token top-8 + renormalized weights (softmax denom cancels:
    //     w_i = exp(l_i - max) / sum_top8 exp(l_j - max)) ---
    {
        const int    t   = threadIdx.x;                 // one token per thread
        const float* row = &logits[t * 129];
        float vals[TOP_K];
        int   ids[TOP_K];
#pragma unroll
        for (int k = 0; k < TOP_K; ++k) { vals[k] = -__builtin_inff(); ids[k] = 0; }

        for (int e = 0; e < NUM_EXPERTS; ++e) {
            float cv = row[e];
            int   ci = e;
#pragma unroll
            for (int p = 0; p < TOP_K; ++p) {
                // strict > keeps the lower index first on ties (jax top_k order)
                if (cv > vals[p]) {
                    float tv = vals[p]; int ti = ids[p];
                    vals[p] = cv; ids[p] = ci;
                    cv = tv; ci = ti;
                }
            }
        }

        const float m = vals[0];           // top-1 == global max logit
        float ex[TOP_K];
        float s = 0.0f;
#pragma unroll
        for (int k = 0; k < TOP_K; ++k) { ex[k] = __expf(vals[k] - m); s += ex[k]; }
        const float inv = 1.0f / s;

        const size_t token = (size_t)blockIdx.x * TOK_PER_WG + t;
#pragma unroll
        for (int k = 0; k < TOP_K; ++k) {
            out_w[token * TOP_K + k] = ex[k] * inv;
            out_i[token * TOP_K + k] = ids[k];
        }
    }
}

extern "C" void kernel_launch(void* const* d_in, const int* in_sizes, int n_in,
                              void* d_out, int out_size, void* d_ws, size_t ws_size,
                              hipStream_t stream) {
    const float* hs = (const float*)d_in[0];   // [131072, 2048] fp32
    const float* gw = (const float*)d_in[1];   // [128, 2048] fp32
    _Float16* bws = (_Float16*)d_ws;           // 512 KB packed B fragments

    float* out_w = (float*)d_out;                                  // [T, 8] fp32
    int*   out_i = (int*)d_out + (size_t)NUM_TOKENS * TOP_K;       // [T, 8] i32

    gate_pack_f16<<<(NUM_EXPERTS * HIDDEN) / 256, 256, 0, stream>>>(gw, bws);
    router_fused<<<NUM_TOKENS / TOK_PER_WG, 128, 0, stream>>>(hs, bws, out_w, out_i);
}
